// SanaMultiscaleLinearAttention_83038897700911
// MI455X (gfx1250) — compile-verified, matrix operand order checked by reading, not run on hardware
//
#include <hip/hip_runtime.h>

// ---------------------------------------------------------------------------
// SANA multiscale linear attention for MI455X (gfx1250), wave32 + WMMA f16.
// GEMMs: v_wmma_f32_16x16x32_f16 with LDS-staged 128x128 tiles.
// A-tile staging uses the Tensor Data Mover (tensor_load_to_lds +
// s_wait_tensorcnt) when the builtin is available; sync staging otherwise.
// ---------------------------------------------------------------------------

typedef __attribute__((ext_vector_type(16))) _Float16 v16h;
typedef __attribute__((ext_vector_type(8)))  float    v8f;

#if defined(__has_builtin)
#if __has_builtin(__builtin_amdgcn_tensor_load_to_lds) && \
    __has_builtin(__builtin_amdgcn_s_wait_tensorcnt)
#define USE_TDM 1
#endif
#endif

#define LDH 40  // LDS row stride in halves (32 data + 8 pad = 64B + 16B pad)

#if defined(USE_TDM)
typedef unsigned int v4ui __attribute__((ext_vector_type(4)));
typedef int          v8si __attribute__((ext_vector_type(8)));
typedef int          v4si __attribute__((ext_vector_type(4)));

// TDM 2D tile load: 128 rows x 32 halves from row-major (rows x K) matrix at
// gptr, into LDS at byte offset lds_off with a 16B pad every 64B (LDH layout).
// D# packing per CDNA5 ISA ch.8 (group0 128b, group1 256b; groups 2/3 zero).
__device__ __forceinline__ void tdm_load_tile_a(unsigned lds_off,
                                                const _Float16* gptr,
                                                int K, int M) {
  unsigned long long ga = (unsigned long long)(uintptr_t)gptr;
  v4ui g0;
  g0.x = 1u;                                          // count=1, user mode
  g0.y = lds_off;                                     // lds_addr [63:32]
  g0.z = (unsigned)(ga & 0xffffffffu);                // global_addr[31:0]
  g0.w = (unsigned)((ga >> 32) & 0x01ffffffu) | (2u << 30);  // [120:96] | type=2
  unsigned w[8];
  w[0] = (1u << 16)      // data_size = 1 -> 2 bytes
       | (1u << 20)      // pad_enable
       | (3u << 22)      // pad_interval: 16 dwords (64B) before each pad
       | (3u << 25);     // pad_amount: 4 dwords (16B)
  w[1] = ((unsigned)K & 0xffffu) << 16;                               // tdim0 lo
  w[2] = (((unsigned)K >> 16) & 0xffffu) | (((unsigned)M & 0xffffu) << 16);
  w[3] = (((unsigned)M >> 16) & 0xffffu) | (32u << 16);               // tile_dim0
  w[4] = 128u;                                       // tile_dim1=128, tile_dim2=0
  w[5] = (unsigned)K;                                // tensor_dim0_stride lo
  w[6] = 0u;
  w[7] = 0u;
  v8si g1;
#pragma unroll
  for (int i = 0; i < 8; ++i) g1[i] = (int)w[i];
  v4si z4 = {0, 0, 0, 0};
#if __clang_major__ >= 23
  v8si z8 = {0, 0, 0, 0, 0, 0, 0, 0};
  __builtin_amdgcn_tensor_load_to_lds(g0, g1, z4, z4, z8, 0);
#else
  __builtin_amdgcn_tensor_load_to_lds(g0, g1, z4, z4, 0);
#endif
}
#endif  // USE_TDM

// Load one 16x32 f16 fragment (A layout; B-tile is stored transposed in LDS
// as [n][k] so it uses the same loader).
// ISA 7.12.2: lane L -> row = L%16; dwords 0..3 hold K = khi+{0..7},
// dwords 4..7 hold K = 16+khi+{0..7}, khi = (L>=16) ? 8 : 0.
__device__ __forceinline__ v16h load_frag16(const _Float16* sm, int row0, int lane) {
  union { v16h v; uint32_t u[8]; } r;
  int row = row0 + (lane & 15);
  int kd  = (lane & 16) ? 4 : 0;
  const uint32_t* p = (const uint32_t*)(sm + row * LDH);
#pragma unroll
  for (int d = 0; d < 4; ++d) r.u[d]     = p[kd + d];
#pragma unroll
  for (int d = 0; d < 4; ++d) r.u[4 + d] = p[8 + kd + d];
  return r.v;
}

// ---------------------------------------------------------------------------
// Tiled WMMA GEMM: D = A(MxK) * Bm(KxN), per-batch offsets on Bm and D.
// Block = 256 threads = 8 waves; tile 128(M) x 128(N) x 32(K).
// Wave w computes a 32x64 sub-tile: 2 (M) x 4 (N) WMMA fragments.
// STORE_T=false: D is f16, row-major (M x N)   [channel-major feature maps]
// STORE_T=true : D is f32, stored transposed as D[n*M + m]
// ---------------------------------------------------------------------------
template <bool STORE_T>
__global__ __launch_bounds__(256) void wmma_gemm_kernel(
    const _Float16* __restrict__ A,
    const _Float16* __restrict__ Bm,
    void* __restrict__ D,
    int M, int N, int K,
    unsigned long long bstrideB, unsigned long long bstrideD,
    int mTiles, int nTiles)
{
  __shared__ _Float16 As[128 * LDH];
  __shared__ _Float16 Bs[128 * LDH];

  int bid = blockIdx.x;
  int nt  = bid % nTiles;
  int mt  = (bid / nTiles) % mTiles;
  int b   = bid / (nTiles * mTiles);
  int m0  = mt * 128, n0 = nt * 128;

  const _Float16* Bb = Bm + (size_t)b * bstrideB;

  int tid  = threadIdx.x;
  int lane = tid & 31;
  int wave = tid >> 5;
  int wm   = (wave >> 1) * 32;   // 0,32,64,96
  int wn   = (wave & 1)  * 64;   // 0,64

#if defined(USE_TDM)
  unsigned as_base = (unsigned)(uintptr_t)(&As[0]);  // LDS byte offset
#endif

  v8f acc[2][4];
#pragma unroll
  for (int i = 0; i < 2; ++i)
#pragma unroll
    for (int j = 0; j < 4; ++j)
#pragma unroll
      for (int r = 0; r < 8; ++r) acc[i][j][r] = 0.0f;

  for (int k0 = 0; k0 < K; k0 += 32) {
#if defined(USE_TDM)
    // --- Stage A tile via Tensor Data Mover (one wave issues the DMA) ---
    if (wave == 0) {
      tdm_load_tile_a(as_base, A + (size_t)m0 * K + k0, K, M);
      __builtin_amdgcn_s_wait_tensorcnt(0);
    }
#else
    // --- Stage A tile: 128 rows x 16 halves (1024 dwords, 4 per thread) ---
#pragma unroll
    for (int i = 0; i < 4; ++i) {
      int idx = tid + 256 * i;
      int row = idx >> 3;
      int cd  = idx & 7;
      uint32_t v = *(const uint32_t*)(A + (size_t)(m0 + row) * K + k0 + cd * 2);
      *(uint32_t*)(As + row * LDH + cd * 2) = v;
    }
#endif
    // --- Stage B tile transposed: Bs[n][k] (2048 dwords read coalesced) ---
#pragma unroll
    for (int i = 0; i < 8; ++i) {
      int idx = tid + 256 * i;
      int kk  = idx >> 6;          // 0..31
      int nd  = idx & 63;          // dword index along n
      uint32_t v = *(const uint32_t*)(Bb + (size_t)(k0 + kk) * N + n0 + nd * 2);
      _Float16 h0 = ((const _Float16*)&v)[0];
      _Float16 h1 = ((const _Float16*)&v)[1];
      Bs[(nd * 2)     * LDH + kk] = h0;
      Bs[(nd * 2 + 1) * LDH + kk] = h1;
    }
    // Prefetch next k-step's B tile rows into cache (global_prefetch_b8).
    if (k0 + 32 < K) {
      const _Float16* nb = Bb + (size_t)(k0 + 32 + (tid >> 3)) * N + n0 + (tid & 7) * 16;
      __builtin_prefetch(nb, 0, 3);
    }
    __syncthreads();

    v16h af[2], bf[4];
#pragma unroll
    for (int i = 0; i < 2; ++i) af[i] = load_frag16(As, wm + 16 * i, lane);
#pragma unroll
    for (int j = 0; j < 4; ++j) bf[j] = load_frag16(Bs, wn + 16 * j, lane);

#pragma unroll
    for (int i = 0; i < 2; ++i)
#pragma unroll
      for (int j = 0; j < 4; ++j)
        acc[i][j] = __builtin_amdgcn_wmma_f32_16x16x32_f16(
            false, af[i], false, bf[j], (short)0, acc[i][j], false, false);
    __syncthreads();
  }

  // --- Epilogue (C/D layout: VGPR r -> M = r + 8*(lane>=16), N = lane%16) ---
  int nlo = lane & 15;
  int mhi = (lane & 16) ? 8 : 0;
  if (!STORE_T) {
    _Float16* Dp = (_Float16*)D + (size_t)b * bstrideD;
#pragma unroll
    for (int i = 0; i < 2; ++i)
#pragma unroll
      for (int j = 0; j < 4; ++j) {
        int mb = m0 + wm + 16 * i + mhi;
        int nn = n0 + wn + 16 * j + nlo;
#pragma unroll
        for (int r = 0; r < 8; ++r)
          Dp[(size_t)(mb + r) * N + nn] = (_Float16)acc[i][j][r];
      }
  } else {
    float* Dp = (float*)D + (size_t)b * bstrideD;
#pragma unroll
    for (int i = 0; i < 2; ++i)
#pragma unroll
      for (int j = 0; j < 4; ++j) {
        int mb = m0 + wm + 16 * i + mhi;
        int nn = n0 + wn + 16 * j + nlo;
        float* dst = Dp + (size_t)nn * M + mb;   // 8 consecutive M per lane
        float4 lo = make_float4(acc[i][j][0], acc[i][j][1], acc[i][j][2], acc[i][j][3]);
        float4 hi = make_float4(acc[i][j][4], acc[i][j][5], acc[i][j][6], acc[i][j][7]);
        *(float4*)(dst)     = lo;
        *(float4*)(dst + 4) = hi;
      }
  }
}

// ---------------------------------------------------------------------------
// fp32 -> f16 elementwise convert (grid-stride)
// ---------------------------------------------------------------------------
__global__ void cvt_f32_f16(const float* __restrict__ in, _Float16* __restrict__ out,
                            unsigned long long n) {
  unsigned long long i = (unsigned long long)blockIdx.x * blockDim.x + threadIdx.x;
  unsigned long long stride = (unsigned long long)gridDim.x * blockDim.x;
  for (; i < n; i += stride) out[i] = (_Float16)in[i];
}

// Pack [Wq;Wk;Wv] (each 1152x1152) into a single 3456x1152 f16 matrix.
__global__ void pack_wqkv(const float* __restrict__ Wq, const float* __restrict__ Wk,
                          const float* __restrict__ Wv, _Float16* __restrict__ out) {
  const unsigned long long n = 3456ull * 1152ull;
  unsigned long long i = (unsigned long long)blockIdx.x * blockDim.x + threadIdx.x;
  unsigned long long stride = (unsigned long long)gridDim.x * blockDim.x;
  for (; i < n; i += stride) {
    unsigned long long row = i / 1152ull;
    unsigned long long col = i - row * 1152ull;
    const float* src = (row < 1152ull) ? Wq : (row < 2304ull) ? Wk : Wv;
    unsigned long long r = row % 1152ull;
    out[i] = (_Float16)src[r * 1152ull + col];
  }
}

// ---------------------------------------------------------------------------
// Fused depthwise 5x5 (pad 2) + grouped 1x1 (8 in -> 8 out per group).
// Reads qkv channels [0,3456) of hs, writes ms into channels [3456,6912).
// Grid: B(2) * groups(432) * spatial tiles(4 of 32x32).
// ---------------------------------------------------------------------------
__global__ __launch_bounds__(256) void conv_ms(
    _Float16* __restrict__ hs,
    const float* __restrict__ w5,    // (3456,25)
    const float* __restrict__ w1)    // (3456,8)
{
  __shared__ _Float16 sm[8 * 36 * 36];
  __shared__ float w5s[8 * 25];
  __shared__ float w1s[8 * 8];

  int bid  = blockIdx.x;
  int tile = bid & 3;
  int g    = (bid >> 2) % 432;
  int b    = (bid >> 2) / 432;
  int ty = (tile >> 1) * 32, tx = (tile & 1) * 32;
  int gbase = g * 8;

  const _Float16* in = hs + ((size_t)b * 6912 + gbase) * 4096;
  int tid = threadIdx.x;

  for (int idx = tid; idx < 8 * 36 * 36; idx += 256) {
    int ci  = idx / 1296;
    int rem = idx - ci * 1296;
    int r = rem / 36, c = rem - (rem / 36) * 36;
    int gy = ty + r - 2, gx = tx + c - 2;
    _Float16 v = (_Float16)0.0f;
    if (gy >= 0 && gy < 64 && gx >= 0 && gx < 64)
      v = in[(size_t)ci * 4096 + gy * 64 + gx];
    sm[idx] = v;
  }
  for (int idx = tid; idx < 200; idx += 256)
    w5s[idx] = w5[(size_t)(gbase + idx / 25) * 25 + (idx % 25)];
  if (tid < 64)
    w1s[tid] = w1[(size_t)(gbase + (tid >> 3)) * 8 + (tid & 7)];
  __syncthreads();

  _Float16* outp = hs + ((size_t)b * 6912 + 3456 + gbase) * 4096;
#pragma unroll
  for (int i = 0; i < 4; ++i) {
    int p  = tid + 256 * i;
    int py = p >> 5, px = p & 31;
    float dw[8];
#pragma unroll
    for (int ci = 0; ci < 8; ++ci) {
      float s = 0.0f;
#pragma unroll
      for (int ky = 0; ky < 5; ++ky)
#pragma unroll
        for (int kx = 0; kx < 5; ++kx)
          s += (float)sm[ci * 1296 + (py + ky) * 36 + (px + kx)] * w5s[ci * 25 + ky * 5 + kx];
      dw[ci] = s;
    }
    int gy = ty + py, gx = tx + px;
#pragma unroll
    for (int co = 0; co < 8; ++co) {
      float s = 0.0f;
#pragma unroll
      for (int ci = 0; ci < 8; ++ci) s += w1s[co * 8 + ci] * dw[ci];
      outp[(size_t)co * 4096 + gy * 64 + gx] = (_Float16)s;
    }
  }
}

// ---------------------------------------------------------------------------
// Linear attention: per (b,g) block. hs layout gives q=ch 24g+0..7,
// k=24g+8..15, v=24g+16..23. scores(9x8) = [v;1] @ relu(k)^T over N=4096,
// out = scores @ relu(q), rows 0..7 normalized by row 8.
// Wave32 shuffle reduce + LDS ds_add_f32 across waves.
// ---------------------------------------------------------------------------
__global__ __launch_bounds__(256) void linattn(const _Float16* __restrict__ hs,
                                               _Float16* __restrict__ attn) {
  int bid = blockIdx.x;
  int g = bid % 288;
  int b = bid / 288;
  const _Float16* base = hs + ((size_t)b * 6912 + (size_t)g * 24) * 4096;

  float acc[72];
#pragma unroll
  for (int i = 0; i < 72; ++i) acc[i] = 0.0f;

  int tid = threadIdx.x;
  for (int n = tid; n < 4096; n += 256) {
    float kv[8], vv[8];
#pragma unroll
    for (int e = 0; e < 8; ++e) {
      float k = (float)base[(size_t)(8 + e) * 4096 + n];
      kv[e] = k > 0.0f ? k : 0.0f;
    }
#pragma unroll
    for (int d = 0; d < 8; ++d) vv[d] = (float)base[(size_t)(16 + d) * 4096 + n];
#pragma unroll
    for (int d = 0; d < 8; ++d)
#pragma unroll
      for (int e = 0; e < 8; ++e) acc[d * 8 + e] += vv[d] * kv[e];
#pragma unroll
    for (int e = 0; e < 8; ++e) acc[64 + e] += kv[e];   // ones row
  }

  __shared__ float red[72];
  for (int i = tid; i < 72; i += 256) red[i] = 0.0f;
  __syncthreads();
#pragma unroll
  for (int i = 0; i < 72; ++i) {
    float s = acc[i];
#pragma unroll
    for (int off = 16; off > 0; off >>= 1) s += __shfl_down(s, off);
    if ((tid & 31) == 0) atomicAdd(&red[i], s);
  }
  __syncthreads();

  float sc[72];
#pragma unroll
  for (int i = 0; i < 72; ++i) sc[i] = red[i];

  _Float16* outp = attn + ((size_t)b * 2304 + (size_t)g * 8) * 4096;
  for (int n = tid; n < 4096; n += 256) {
    float qv[8];
#pragma unroll
    for (int e = 0; e < 8; ++e) {
      float q = (float)base[(size_t)e * 4096 + n];
      qv[e] = q > 0.0f ? q : 0.0f;
    }
    float o[9];
#pragma unroll
    for (int d = 0; d < 9; ++d) {
      float s = 0.0f;
#pragma unroll
      for (int e = 0; e < 8; ++e) s += sc[d * 8 + e] * qv[e];
      o[d] = s;
    }
    float inv = 1.0f / (o[8] + 1e-15f);
#pragma unroll
    for (int d = 0; d < 8; ++d)
      outp[(size_t)d * 4096 + n] = (_Float16)(o[d] * inv);
  }
}

// ---------------------------------------------------------------------------
// RMSNorm over channels: input proj_t (B, N, 1152) fp32 (contiguous rows),
// output NCHW (B, 1152, 4096) fp32.
// ---------------------------------------------------------------------------
__global__ __launch_bounds__(256) void rmsnorm(const float* __restrict__ projt,
                                               const float* __restrict__ rw,
                                               const float* __restrict__ rb,
                                               float* __restrict__ out) {
  int bid = blockIdx.x;
  int n = bid % 4096;
  int b = bid / 4096;
  const float* row = projt + ((size_t)b * 4096 + n) * 1152;
  int tid = threadIdx.x;

  float ss = 0.0f;
  for (int c = tid; c < 1152; c += 256) { float v = row[c]; ss += v * v; }
#pragma unroll
  for (int off = 16; off > 0; off >>= 1) ss += __shfl_down(ss, off);

  __shared__ float r8[8];
  if ((tid & 31) == 0) r8[tid >> 5] = ss;
  __syncthreads();
  if (tid == 0) {
    float t = 0.0f;
#pragma unroll
    for (int i = 0; i < 8; ++i) t += r8[i];
    r8[0] = t;
  }
  __syncthreads();
  float scale = rsqrtf(r8[0] / 1152.0f + 1e-5f);

  float* ob = out + (size_t)b * 1152 * 4096;
  for (int c = tid; c < 1152; c += 256)
    ob[(size_t)c * 4096 + n] = row[c] * scale * rw[c] + rb[c];
}

// ---------------------------------------------------------------------------
extern "C" void kernel_launch(void* const* d_in, const int* in_sizes, int n_in,
                              void* d_out, int out_size, void* d_ws, size_t ws_size,
                              hipStream_t stream) {
  const float* x    = (const float*)d_in[0];
  const float* Wq   = (const float*)d_in[1];
  const float* Wk   = (const float*)d_in[2];
  const float* Wv   = (const float*)d_in[3];
  const float* w5   = (const float*)d_in[4];
  const float* w1   = (const float*)d_in[5];
  const float* Wout = (const float*)d_in[6];
  const float* rmsw = (const float*)d_in[7];
  const float* rmsb = (const float*)d_in[8];
  float* out = (float*)d_out;
  (void)in_sizes; (void)n_in; (void)out_size; (void)ws_size;

  char* ws = (char*)d_ws;
  size_t off = 0;
  auto carve = [&](size_t bytes) -> char* {
    char* p = ws + off;
    off = (off + bytes + 255) & ~(size_t)255;
    return p;
  };
  _Float16* x16    = (_Float16*)carve(2ull * 1152 * 4096 * 2);  // X in f16 (K x N per b)
  _Float16* wqkv   = (_Float16*)carve(3456ull * 1152 * 2);      // [Wq;Wk;Wv] f16
  _Float16* wout16 = (_Float16*)carve(1152ull * 2304 * 2);      // Wout f16
  _Float16* hs     = (_Float16*)carve(2ull * 6912 * 4096 * 2);  // qkv | ms, channel-major
  _Float16* attn   = (_Float16*)carve(2ull * 2304 * 4096 * 2);  // attention out, channel-major
  float*    projt  = (float*)carve(2ull * 4096 * 1152 * 4);     // projection out, (B,N,C)

  // 0) precision packing
  cvt_f32_f16<<<4096, 256, 0, stream>>>(x, x16, 2ull * 1152 * 4096);
  pack_wqkv<<<2048, 256, 0, stream>>>(Wq, Wk, Wv, wqkv);
  cvt_f32_f16<<<2048, 256, 0, stream>>>(Wout, wout16, 1152ull * 2304);

  // 1) QKV GEMM: hs[b, 0:3456, :] = Wqkv @ X_b            (WMMA f16)
  wmma_gemm_kernel<false><<<2 * 27 * 32, 256, 0, stream>>>(
      wqkv, x16, hs, 3456, 4096, 1152,
      1152ull * 4096, 6912ull * 4096, 27, 32);

  // 2) multiscale branch: depthwise 5x5 + grouped 1x1 -> hs[b, 3456:6912, :]
  conv_ms<<<2 * 432 * 4, 256, 0, stream>>>(hs, w5, w1);

  // 3) linear attention per (b, g)
  linattn<<<2 * 288, 256, 0, stream>>>(hs, attn);

  // 4) output projection: projt[b, n, :] = (Wout @ attn_b)^T  (WMMA f16)
  wmma_gemm_kernel<true><<<2 * 9 * 32, 256, 0, stream>>>(
      wout16, attn, projt, 1152, 4096, 2304,
      2304ull * 4096, 4096ull * 1152, 9, 32);

  // 5) RMSNorm + transpose to NCHW
  rmsnorm<<<2 * 4096, 256, 0, stream>>>(projt, rmsw, rmsb, out);
}
